// TransMILMSPFPEG_59983513256395
// MI455X (gfx1250) — matrix operand order repaired and checked by reading, not compiled, not run on hardware
//
#include <hip/hip_runtime.h>

// ---------------------------------------------------------------------------
// MI455X (gfx1250) implementation of TransMIL forward.
// Heavy GEMMs -> v_wmma_f32_16x16x32_bf16 (wave32). The 64-wide B panel is
// staged into LDS via global_load_async_to_lds_b128 (ASYNCcnt, double-buffered
// ping-pong, no register copies); each wave owns two 16-row tiles so every
// staged panel feeds 8 back-to-back WMMAs per wave.
// Whole working set (~180MB) is L2-resident (192MB L2).
// ---------------------------------------------------------------------------

typedef __bf16 v16bf __attribute__((ext_vector_type(16)));
typedef __bf16 v8bf  __attribute__((ext_vector_type(8)));
typedef float  v8f   __attribute__((ext_vector_type(8)));

#define NTOK  16384
#define CDIM  512
#define NHEAD 8
#define DHEAD 64
#define NLM   256      // landmarks
#define LGRP  64       // tokens per landmark group (NTOK / NLM)
#define HIMG  128      // image side for FPEG (128*128 == NTOK)

// gfx1250 async copy: global -> LDS, 16B per lane, tracked by ASYNCcnt.
__device__ __forceinline__ void async_g2l_b128(unsigned ldsOff, const void* gptr)
{
    unsigned long ga = (unsigned long)gptr;
    asm volatile("global_load_async_to_lds_b128 %0, %1, off"
                 :: "v"(ldsOff), "v"(ga) : "memory");
}
__device__ __forceinline__ void wait_async0()
{
    asm volatile("s_wait_asynccnt 0x0" ::: "memory");
}
__device__ __forceinline__ unsigned lds_off(const void* p)
{
    return (unsigned)(unsigned long)p;   // low 32 bits of generic shared addr = LDS offset
}

// ===========================================================================
// WMMA bf16 GEMM:  C[M,N] = alpha * A[M,K] x Bt[N,K]^T  (+bias,+add,relu)
// A row-major bf16 (lda), Bt row-major bf16 (ldb) == B transposed.
// Block = 128 threads (4 waves) computing a 128x64 tile: wave w owns rows
// [bm*128 + w*32, +32) as two 16-row tiles; all waves share the 64-wide B
// panel staged in LDS by async copies (ping-pong double buffer). A fragments
// ping-pong in registers (loop unrolled by 2; nk even at all call sites).
// Requires M%128==0, N%64==0, K%64==0 (all call sites satisfy this).
// ===========================================================================
#define BP_PAD 40   // bf16 elems per LDS row (80B = 20 banks: conflict-free, 16B aligned)

__global__ __launch_bounds__(128)
void k_wmma_gemm(const __bf16* __restrict__ A, const __bf16* __restrict__ Bt,
                 float* __restrict__ C, __bf16* __restrict__ Cbf,
                 const float* __restrict__ bias, const float* __restrict__ addsrc,
                 int M, int N, int K, int lda, int ldb, int ldc, int ldadd,
                 long sA, long sB, long sC, float alpha, int relu)
{
    __shared__ __bf16 Bs[2][64][BP_PAD];

    const int lane   = threadIdx.x & 31;
    const int wv     = threadIdx.x >> 5;
    const int tilesN = N >> 6;
    const int bm     = blockIdx.x / tilesN;
    const int bn     = blockIdx.x - bm * tilesN;

    const long b = blockIdx.z;
    A  += b * sA;
    Bt += b * sB;

    const int half = lane >> 4;               // 0 or 1
    const int l15  = lane & 15;
    const int ka   = half * 8;                // A frag: e<8 -> ka+e ; e>=8 -> 16+ka+(e-8)
    const int kb   = half * 8;                // B frag: e<8 -> kb+e ; e>=8 -> 16+kb+(e-8)

    const int mrow = bm * 128 + wv * 32 + l15;            // lane row of tile 0
    const __bf16* __restrict__ ap0 = A + (long)mrow * lda + ka;
    const __bf16* __restrict__ ap1 = ap0 + (long)16 * lda; // tile 1 (+16 rows)

    // ---- B panel staging: 64 rows x 32 k of bf16 (4KB) per k-step ----
    const int  sr = threadIdx.x >> 1;          // staging row 0..63
    const int  sc = (threadIdx.x & 1) * 16;    // staging col base 0 / 16
    const __bf16* __restrict__ gB = Bt + (long)(bn * 64 + sr) * ldb + sc;

    auto stage = [&](int buf, int k0) {
        const __bf16* g = gB + k0;
        unsigned d = lds_off(&Bs[buf][sr][sc]);
        async_g2l_b128(d,      g);
        async_g2l_b128(d + 16, g + 8);
    };

    union AF { v16bf v; v8bf h[2]; };
    auto loadA = [&](AF* a, int kk) {
        const __bf16* p0 = ap0 + ((long)kk << 5);
        const __bf16* p1 = ap1 + ((long)kk << 5);
        a[0].h[0] = *(const v8bf*)(p0);
        a[0].h[1] = *(const v8bf*)(p0 + 16);
        a[1].h[0] = *(const v8bf*)(p1);
        a[1].h[1] = *(const v8bf*)(p1 + 16);
    };

    v8f zero = {0.f,0.f,0.f,0.f,0.f,0.f,0.f,0.f};
    v8f acc[2][4];
#pragma unroll
    for (int i = 0; i < 2; ++i)
#pragma unroll
        for (int j = 0; j < 4; ++j) acc[i][j] = zero;

    // All 8 DS loads first (one clause, one wait), then 8 back-to-back WMMAs.
    auto compute = [&](int buf, AF* a) {
        AF bu[4];
#pragma unroll
        for (int j = 0; j < 4; ++j) {
            const __bf16* bs = &Bs[buf][j * 16 + l15][kb];
            bu[j].h[0] = *(const v8bf*)(bs);
            bu[j].h[1] = *(const v8bf*)(bs + 16);
        }
#pragma unroll
        for (int i = 0; i < 2; ++i)
#pragma unroll
            for (int j = 0; j < 4; ++j)
                acc[i][j] = __builtin_amdgcn_wmma_f32_16x16x32_bf16(
                    false, a[i].v, false, bu[j].v, (short)0, acc[i][j], false, false);
    };

    const int nk = K >> 5;   // even at every call site
    AF aP[2][2];

    stage(0, 0);
    loadA(aP[0], 0);
    wait_async0();
    __syncthreads();

    for (int kk = 0; kk < nk; kk += 2) {
        stage(1, (kk + 1) << 5);            // async: overlaps WMMAs
        loadA(aP[1], kk + 1);               // reg ping-pong A
        compute(0, aP[0]);
        wait_async0();                      // buf1 landed
        __syncthreads();                    // all waves done reading buf0

        if (kk + 2 < nk) {
            stage(0, (kk + 2) << 5);
            loadA(aP[0], kk + 2);
        }
        compute(1, aP[1]);
        wait_async0();
        __syncthreads();
    }

    // C/D layout: n = lane&15 ; VGPR r -> m = r + 8*(lane>=16)
    const int  rbase = half * 8;
    const long cb    = b * sC;
#pragma unroll
    for (int i = 0; i < 2; ++i) {
#pragma unroll
        for (int j = 0; j < 4; ++j) {
            const int   cn = bn * 64 + j * 16 + l15;
            const float bi = bias ? bias[cn] : 0.f;
#pragma unroll
            for (int r = 0; r < 8; ++r) {
                const int cm = bm * 128 + wv * 32 + i * 16 + rbase + r;
                float v = acc[i][j][r] * alpha + bi;
                if (addsrc) v += addsrc[cb + (long)cm * ldadd + cn];
                if (relu)   v = fmaxf(v, 0.f);
                if (Cbf) Cbf[cb + (long)cm * ldc + cn] = (__bf16)v;
                else     C  [cb + (long)cm * ldc + cn] = v;
            }
        }
    }
}

// ===========================================================================
// Small utility kernels
// ===========================================================================
__global__ void k_f32_to_bf16(const float* __restrict__ in, __bf16* __restrict__ out, long n)
{
    long i = (long)blockIdx.x * blockDim.x + threadIdx.x;
    if (i < n) out[i] = (__bf16)in[i];
}

__global__ void k_add_f32(float* __restrict__ out, const float* __restrict__ in, long n)
{
    long i = (long)blockIdx.x * blockDim.x + threadIdx.x;
    if (i < n) out[i] += in[i];
}

// out[r*C + c] = in[c*ldIn + r] * (r<scaleRows ? scale : 1)   (batched via grid.z)
__global__ void k_transpose_f2b(const float* __restrict__ in, __bf16* __restrict__ out,
                                int R, int C, int ldIn, float scale, int scaleRows,
                                long sIn, long sOut)
{
    long idx = (long)blockIdx.x * blockDim.x + threadIdx.x;
    if (idx >= (long)R * C) return;
    int  r = (int)(idx / C), c = (int)(idx - (long)r * C);
    long b = blockIdx.z;
    float v = in[b * sIn + (long)c * ldIn + r];
    if (r < scaleRows) v *= scale;
    out[b * sOut + (long)r * C + c] = (__bf16)v;
}

__global__ void k_transpose_b2b(const __bf16* __restrict__ in, __bf16* __restrict__ out,
                                int R, int C, int ldIn)
{
    long idx = (long)blockIdx.x * blockDim.x + threadIdx.x;
    if (idx >= (long)R * C) return;
    int r = (int)(idx / C), c = (int)(idx - (long)r * C);
    out[(long)r * C + c] = in[(long)c * ldIn + r];
}

// LayerNorm over rows of length CDIM, one wave per row (wave32), 4 waves/block.
__global__ __launch_bounds__(128)
void k_layernorm(const float* __restrict__ x, const float* __restrict__ w,
                 const float* __restrict__ b, __bf16* __restrict__ outb,
                 float* __restrict__ outf, int n)
{
    int row = blockIdx.x * 4 + (threadIdx.x >> 5);
    if (row >= n) return;
    int lane = threadIdx.x & 31;
    const float* xr = x + (long)row * CDIM;
    float s = 0.f, ss = 0.f;
    for (int c = lane; c < CDIM; c += 32) { float v = xr[c]; s += v; ss += v * v; }
    for (int o = 16; o; o >>= 1) { s += __shfl_xor(s, o, 32); ss += __shfl_xor(ss, o, 32); }
    float mu = s * (1.f / CDIM);
    float var = ss * (1.f / CDIM) - mu * mu;
    float rs = rsqrtf(var + 1e-5f);
    for (int c = lane; c < CDIM; c += 32) {
        float v = (xr[c] - mu) * rs * w[c] + b[c];
        if (outb) outb[(long)row * CDIM + c] = (__bf16)v;
        if (outf) outf[(long)row * CDIM + c] = v;
    }
}

// Row softmax, row length L, one block (256 threads) per row. Output bf16/f32.
__global__ __launch_bounds__(256)
void k_softmax(const float* __restrict__ X, __bf16* __restrict__ ob,
               float* __restrict__ of, int L)
{
    long row = blockIdx.x;
    const float* xr = X + row * (long)L;
    __shared__ float red[256];
    int t = threadIdx.x;
    float mx = -3.0e38f;
    for (int i = t; i < L; i += 256) mx = fmaxf(mx, xr[i]);
    red[t] = mx; __syncthreads();
    for (int o = 128; o; o >>= 1) { if (t < o) red[t] = fmaxf(red[t], red[t + o]); __syncthreads(); }
    mx = red[0]; __syncthreads();
    float s = 0.f;
    for (int i = t; i < L; i += 256) s += __expf(xr[i] - mx);
    red[t] = s; __syncthreads();
    for (int o = 128; o; o >>= 1) { if (t < o) red[t] += red[t + o]; __syncthreads(); }
    float inv = 1.f / red[0];
    for (int i = t; i < L; i += 256) {
        float v = __expf(xr[i] - mx) * inv;
        if (ob) ob[row * (long)L + i] = (__bf16)v;
        if (of) of[row * (long)L + i] = v;
    }
}

// Landmark mean-pool: dst[h][j][d] = mean_{r<64} src[(j*64+r)*ld + colOff + h*64 + d]
__global__ void k_landmark(const __bf16* __restrict__ src, int ld, int colOff,
                           __bf16* __restrict__ dst)
{
    long idx = (long)blockIdx.x * blockDim.x + threadIdx.x;
    if (idx >= (long)NHEAD * NLM * DHEAD) return;
    int h = (int)(idx >> 14), rem = (int)(idx & 16383);
    int j = rem >> 6, d = rem & 63;
    const __bf16* p = src + (long)(j * LGRP) * ld + colOff + h * DHEAD + d;
    float s = 0.f;
    for (int r = 0; r < LGRP; ++r) s += (float)p[(long)r * ld];
    dst[idx] = (__bf16)(s * (1.f / LGRP));
}

__global__ void k_init2(unsigned* p) { if (threadIdx.x < 2) p[threadIdx.x] = 0u; }

// max over (h,r) of sum_t |X[h*65536 + r*rowOff + t*elemStride]| ; 2048 blocks x 256
__global__ __launch_bounds__(256)
void k_abssum_max(const float* __restrict__ X, unsigned* outMax, int rowOff, int elemStride)
{
    int bi = blockIdx.x, h = bi >> 8, r = bi & 255;
    const float* base = X + (long)h * 65536 + (long)r * rowOff;
    __shared__ float red[256];
    red[threadIdx.x] = fabsf(base[(long)threadIdx.x * elemStride]);
    __syncthreads();
    for (int o = 128; o; o >>= 1) { if (threadIdx.x < o) red[threadIdx.x] += red[threadIdx.x + o]; __syncthreads(); }
    if (threadIdx.x == 0) atomicMax(outMax, __float_as_uint(red[0]));
}

// z[h][r][c] = a2[h][c][r] / (s1*s2)
__global__ void k_zinit(const float* __restrict__ a2, float* __restrict__ z,
                        const unsigned* __restrict__ mx)
{
    long idx = (long)blockIdx.x * blockDim.x + threadIdx.x;
    if (idx >= (long)NHEAD * 65536) return;
    float rcp = 1.f / (__uint_as_float(mx[0]) * __uint_as_float(mx[1]));
    int h = (int)(idx >> 16), r = (int)((idx >> 8) & 255), c = (int)(idx & 255);
    z[idx] = a2[(long)h * 65536 + (long)c * 256 + r] * rcp;
}

// y = cv*I - x   (batched [8][256][256] flat)
__global__ void k_axpyI(const float* __restrict__ x, float* __restrict__ y, float cv)
{
    long idx = (long)blockIdx.x * blockDim.x + threadIdx.x;
    if (idx >= (long)NHEAD * 65536) return;
    int r = (int)((idx >> 8) & 255), c = (int)(idx & 255);
    y[idx] = (r == c ? cv : 0.f) - x[idx];
}

// Residual depthwise conv over sequence (kernel 33) on v, added into attn-out.
__global__ void k_resconv(const __bf16* __restrict__ qkv, const float* __restrict__ rw,
                          float* __restrict__ out, int n)
{
    long idx = (long)blockIdx.x * blockDim.x + threadIdx.x;
    if (idx >= (long)n * CDIM) return;
    int i = (int)(idx >> 9), c = (int)(idx & 511), h = c >> 6;
    float s = 0.f;
    for (int t = 0; t < 33; ++t) {
        int ii = i + t - 16;
        if (ii >= 0 && ii < n)
            s += rw[h * 33 + t] * (float)qkv[(long)ii * (3 * CDIM) + 2 * CDIM + c];
    }
    out[idx] += s;
}

// FPEG: sum of three depthwise 2D convs (+bias each), optional relu.
__device__ __forceinline__ float dwtap(const float* __restrict__ in,
                                       const float* __restrict__ w,
                                       int k, int c, int y, int x)
{
    float s = 0.f; int hk = k >> 1;
    for (int dy = 0; dy < k; ++dy) {
        int yy = y + dy - hk; if ((unsigned)yy >= (unsigned)HIMG) continue;
        for (int dx = 0; dx < k; ++dx) {
            int xx = x + dx - hk; if ((unsigned)xx >= (unsigned)HIMG) continue;
            s += w[c * k * k + dy * k + dx] * in[((long)(yy * HIMG + xx)) * CDIM + c];
        }
    }
    return s;
}

__global__ __launch_bounds__(256)
void k_fpeg3(const float* __restrict__ in, float* __restrict__ out,
             const float* w0, const float* b0, int k0,
             const float* w1, const float* b1, int k1,
             const float* w2, const float* b2, int k2, int relu)
{
    long idx = (long)blockIdx.x * blockDim.x + threadIdx.x;
    if (idx >= (long)NTOK * CDIM) return;
    int pix = (int)(idx >> 9), c = (int)(idx & 511);
    int y = pix >> 7, x = pix & 127;
    float s = b0[c] + b1[c] + b2[c];
    s += dwtap(in, w0, k0, c, y, x);
    s += dwtap(in, w1, k1, c, y, x);
    s += dwtap(in, w2, k2, c, y, x);
    if (relu) s = fmaxf(s, 0.f);
    out[idx] = s;
}

// fc2 logits: one wave per row -> [n,2]
__global__ __launch_bounds__(128)
void k_fc2(const float* __restrict__ hln, const float* __restrict__ w,
           const float* __restrict__ b, float* __restrict__ logits, int n)
{
    int row = blockIdx.x * 4 + (threadIdx.x >> 5);
    if (row >= n) return;
    int lane = threadIdx.x & 31;
    float s0 = 0.f, s1 = 0.f;
    const float* xr = hln + (long)row * CDIM;
    for (int c = lane; c < CDIM; c += 32) {
        float v = xr[c];
        s0 += v * w[c * 2 + 0];
        s1 += v * w[c * 2 + 1];
    }
    for (int o = 16; o; o >>= 1) { s0 += __shfl_xor(s0, o, 32); s1 += __shfl_xor(s1, o, 32); }
    if (lane == 0) { logits[row * 2 + 0] = s0 + b[0]; logits[row * 2 + 1] = s1 + b[1]; }
}

// argmax_i (l1-l0) -> select logits, softmax(2), argmax(2). Writes 5 floats.
__global__ __launch_bounds__(256)
void k_final(const float* __restrict__ logits, float* __restrict__ out, int n)
{
    __shared__ float bv[256];
    __shared__ int   bi[256];
    int t = threadIdx.x;
    float best = -3.0e38f; int bidx = 0;
    for (int i = t; i < n; i += 256) {
        float d = logits[2 * i + 1] - logits[2 * i];
        if (d > best) { best = d; bidx = i; }
    }
    bv[t] = best; bi[t] = bidx; __syncthreads();
    for (int o = 128; o; o >>= 1) {
        if (t < o) {
            if (bv[t + o] > bv[t] || (bv[t + o] == bv[t] && bi[t + o] < bi[t])) {
                bv[t] = bv[t + o]; bi[t] = bi[t + o];
            }
        }
        __syncthreads();
    }
    if (t == 0) {
        int ind = bi[0];
        float l0 = logits[2 * ind], l1 = logits[2 * ind + 1];
        float m = fmaxf(l0, l1);
        float e0 = __expf(l0 - m), e1 = __expf(l1 - m);
        float is = 1.f / (e0 + e1);
        out[0] = l0; out[1] = l1; out[2] = e0 * is; out[3] = e1 * is;
        out[4] = (l1 > l0) ? 1.f : 0.f;
    }
}

// ===========================================================================
// Host-side orchestration
// ===========================================================================
extern "C" void kernel_launch(void* const* d_in, const int* in_sizes, int n_in,
                              void* d_out, int out_size, void* d_ws, size_t ws_size,
                              hipStream_t stream)
{
    (void)in_sizes; (void)n_in; (void)out_size; (void)ws_size;

    const float* data   = (const float*)d_in[0];
    const float* fc1_w  = (const float*)d_in[1];
    const float* fc1_b  = (const float*)d_in[2];
    const float* norm_w = (const float*)d_in[3];
    const float* norm_b = (const float*)d_in[4];
    const float* fc2_w  = (const float*)d_in[5];
    const float* fc2_b  = (const float*)d_in[6];
    const float* ln_w[2]  = {(const float*)d_in[7],  (const float*)d_in[13]};
    const float* ln_b[2]  = {(const float*)d_in[8],  (const float*)d_in[14]};
    const float* qkv_w[2] = {(const float*)d_in[9],  (const float*)d_in[15]};
    const float* ao_w[2]  = {(const float*)d_in[10], (const float*)d_in[16]};
    const float* ao_b[2]  = {(const float*)d_in[11], (const float*)d_in[17]};
    const float* res_w[2] = {(const float*)d_in[12], (const float*)d_in[18]};
    const float* pj[16];
    for (int i = 0; i < 16; ++i) pj[i] = (const float*)d_in[19 + i];

    // ---- workspace layout (aliased regions noted; everything L2-resident) ----
    char* ws = (char*)d_ws;
    size_t off = 0;
    auto alloc = [&](size_t bytes) -> char* {
        char* p = ws + off; off = (off + bytes + 255) & ~(size_t)255; return p;
    };
    float*  hbuf  = (float*)  alloc((size_t)NTOK * CDIM * 4);
    __bf16* xln   = (__bf16*) alloc((size_t)NTOK * CDIM * 2);        // also attn_out bf16
    __bf16* qkvb  = (__bf16*) alloc((size_t)NTOK * 3 * CDIM * 2);    // also FPEG buf A (f32)
    float*  attn  = (float*)  alloc((size_t)NTOK * CDIM * 4);        // also FPEG buf B / hln
    float*  scor  = (float*)  alloc((size_t)NTOK * NLM * 4);         // also logits_all
    __bf16* scorb = (__bf16*) alloc((size_t)NTOK * NLM * 2);
    __bf16* vT    = (__bf16*) alloc((size_t)DHEAD * NTOK * 2);
    __bf16* fc1T  = (__bf16*) alloc((size_t)512 * 512 * 2);
    __bf16* qkvT  = (__bf16*) alloc((size_t)1536 * 512 * 2);
    __bf16* aoT   = (__bf16*) alloc((size_t)512 * 512 * 2);
    __bf16* qlb   = (__bf16*) alloc((size_t)NHEAD * NLM * DHEAD * 2);
    __bf16* klb   = (__bf16*) alloc((size_t)NHEAD * NLM * DHEAD * 2);
    float*  a2f   = (float*)  alloc((size_t)NHEAD * 65536 * 4);
    __bf16* a2b   = (__bf16*) alloc((size_t)NHEAD * 65536 * 2);
    float*  zf    = (float*)  alloc((size_t)NHEAD * 65536 * 4);
    __bf16* zb    = (__bf16*) alloc((size_t)NHEAD * 65536 * 2);
    float*  xzf   = (float*)  alloc((size_t)NHEAD * 65536 * 4);
    __bf16* xzb   = (__bf16*) alloc((size_t)NHEAD * 65536 * 2);
    float*  yf    = (float*)  alloc((size_t)NHEAD * 65536 * 4);
    float*  tf    = (float*)  alloc((size_t)NHEAD * 65536 * 4);
    __bf16* bfT0  = (__bf16*) alloc((size_t)NHEAD * 65536 * 2);
    float*  a3vf  = (float*)  alloc((size_t)NLM * DHEAD * 4);
    float*  Wf    = (float*)  alloc((size_t)NLM * DHEAD * 4);
    __bf16* WTb   = (__bf16*) alloc((size_t)DHEAD * NLM * 2);
    unsigned* mxu = (unsigned*)alloc(256);
    float* fpA = (float*)qkvb;    // alias: free during FPEG
    float* fpB = attn;            // alias: free during FPEG
    float* hln = attn;            // alias: free after layer 2

    auto nb = [](long tot) { return (unsigned)((tot + 255) / 256); };

    auto gemm = [&](const __bf16* Am, const __bf16* Bt, float* C, __bf16* Cb,
                    const float* bias, const float* add,
                    int M, int N, int K, int lda, int ldb, int ldc, int ldadd,
                    long sA, long sB, long sC, float alpha, int relu, int batch) {
        int blocks = (M / 128) * (N / 64);      // 128x64 tile per block, no tail
        dim3 g(blocks, 1, batch);
        k_wmma_gemm<<<g, dim3(128), 0, stream>>>(Am, Bt, C, Cb, bias, add,
                                                 M, N, K, lda, ldb, ldc, ldadd,
                                                 sA, sB, sC, alpha, relu);
    };

    // ---- fc1: h = relu(data @ fc1_w + b) ----
    k_transpose_f2b<<<nb(512 * 512), 256, 0, stream>>>(fc1_w, fc1T, 512, 512, 512, 1.f, 0, 0, 0);
    k_f32_to_bf16<<<nb((long)NTOK * CDIM), 256, 0, stream>>>(data, xln, (long)NTOK * CDIM);
    gemm(xln, fc1T, hbuf, nullptr, fc1_b, nullptr,
         NTOK, 512, 512, 512, 512, 512, 0, 0, 0, 0, 1.f, 1, 1);

    // ---- one TransLayer (Nystrom attention) ----
    auto layer = [&](int li) {
        // LN -> bf16
        k_layernorm<<<(NTOK + 3) / 4, 128, 0, stream>>>(hbuf, ln_w[li], ln_b[li], xln, nullptr, NTOK);
        // qkv weights transposed; q-scale (DH^-0.5 = 0.125) folded into rows [0,512)
        k_transpose_f2b<<<nb((long)1536 * 512), 256, 0, stream>>>(qkv_w[li], qkvT, 1536, 512, 1536, 0.125f, 512, 0, 0);
        gemm(xln, qkvT, nullptr, qkvb, nullptr, nullptr,
             NTOK, 1536, 512, 512, 512, 1536, 0, 0, 0, 0, 1.f, 0, 1);
        // landmarks
        k_landmark<<<nb((long)NHEAD * NLM * DHEAD), 256, 0, stream>>>(qkvb, 1536, 0, qlb);
        k_landmark<<<nb((long)NHEAD * NLM * DHEAD), 256, 0, stream>>>(qkvb, 1536, 512, klb);
        // a2 = softmax(q_l k_l^T)  (batched over heads)
        gemm(qlb, klb, xzf, nullptr, nullptr, nullptr,
             NLM, NLM, DHEAD, DHEAD, DHEAD, NLM, 0, 16384, 16384, 65536, 1.f, 0, NHEAD);
        k_softmax<<<NHEAD * NLM, 256, 0, stream>>>(xzf, a2b, a2f, NLM);
        // Moore-Penrose pinv init: z = a2^T / (max rowsum * max colsum)
        k_init2<<<1, 32, 0, stream>>>(mxu);
        k_abssum_max<<<NHEAD * NLM, 256, 0, stream>>>(a2f, mxu + 0, 256, 1);
        k_abssum_max<<<NHEAD * NLM, 256, 0, stream>>>(a2f, mxu + 1, 1, 256);
        k_zinit<<<nb((long)NHEAD * 65536), 256, 0, stream>>>(a2f, zf, mxu);
        for (int it = 0; it < 6; ++it) {
            dim3 gt(nb(65536), 1, NHEAD);
            // xz = a2 @ z
            k_transpose_f2b<<<gt, 256, 0, stream>>>(zf, bfT0, 256, 256, 256, 1.f, 0, 65536, 65536);
            gemm(a2b, bfT0, xzf, nullptr, nullptr, nullptr,
                 NLM, NLM, NLM, NLM, NLM, NLM, 0, 65536, 65536, 65536, 1.f, 0, NHEAD);
            k_f32_to_bf16<<<nb((long)NHEAD * 65536), 256, 0, stream>>>(xzf, xzb, (long)NHEAD * 65536);
            // t = xz @ (7I - xz)
            k_axpyI<<<nb((long)NHEAD * 65536), 256, 0, stream>>>(xzf, yf, 7.f);
            k_transpose_f2b<<<gt, 256, 0, stream>>>(yf, bfT0, 256, 256, 256, 1.f, 0, 65536, 65536);
            gemm(xzb, bfT0, tf, nullptr, nullptr, nullptr,
                 NLM, NLM, NLM, NLM, NLM, NLM, 0, 65536, 65536, 65536, 1.f, 0, NHEAD);
            // t = xz @ (15I - t)
            k_axpyI<<<nb((long)NHEAD * 65536), 256, 0, stream>>>(tf, yf, 15.f);
            k_transpose_f2b<<<gt, 256, 0, stream>>>(yf, bfT0, 256, 256, 256, 1.f, 0, 65536, 65536);
            gemm(xzb, bfT0, tf, nullptr, nullptr, nullptr,
                 NLM, NLM, NLM, NLM, NLM, NLM, 0, 65536, 65536, 65536, 1.f, 0, NHEAD);
            // z = 0.25 * z @ (13I - t)
            k_axpyI<<<nb((long)NHEAD * 65536), 256, 0, stream>>>(tf, yf, 13.f);
            k_transpose_f2b<<<gt, 256, 0, stream>>>(yf, bfT0, 256, 256, 256, 1.f, 0, 65536, 65536);
            k_f32_to_bf16<<<nb((long)NHEAD * 65536), 256, 0, stream>>>(zf, zb, (long)NHEAD * 65536);
            gemm(zb, bfT0, zf, nullptr, nullptr, nullptr,
                 NLM, NLM, NLM, NLM, NLM, NLM, 0, 65536, 65536, 65536, 0.25f, 0, NHEAD);
        }
        k_f32_to_bf16<<<nb((long)NHEAD * 65536), 256, 0, stream>>>(zf, zb, (long)NHEAD * 65536);

        // per-head: out = softmax(q k_l^T) @ (z @ (softmax(q_l k^T) @ v))
        for (int h = 0; h < NHEAD; ++h) {
            // P3 = softmax(q_l @ k^T)  [256, 16384]
            gemm(qlb + (long)h * 16384, qkvb + 512 + h * DHEAD, scor, nullptr, nullptr, nullptr,
                 NLM, NTOK, DHEAD, DHEAD, 1536, NTOK, 0, 0, 0, 0, 1.f, 0, 1);
            k_softmax<<<NLM, 256, 0, stream>>>(scor, scorb, nullptr, NTOK);
            // vT [64, 16384]
            k_transpose_b2b<<<nb((long)DHEAD * NTOK), 256, 0, stream>>>(qkvb + 1024 + h * DHEAD, vT, DHEAD, NTOK, 1536);
            // a3v = P3 @ v   [256, 64]
            gemm(scorb, vT, a3vf, nullptr, nullptr, nullptr,
                 NLM, DHEAD, NTOK, NTOK, NTOK, DHEAD, 0, 0, 0, 0, 1.f, 0, 1);
            // W = z_h @ a3v  [256, 64]
            k_transpose_f2b<<<nb((long)DHEAD * NLM), 256, 0, stream>>>(a3vf, bfT0, DHEAD, NLM, DHEAD, 1.f, 0, 0, 0);
            gemm(zb + (long)h * 65536, bfT0, Wf, nullptr, nullptr, nullptr,
                 NLM, DHEAD, NLM, NLM, NLM, DHEAD, 0, 0, 0, 0, 1.f, 0, 1);
            k_transpose_f2b<<<nb((long)DHEAD * NLM), 256, 0, stream>>>(Wf, WTb, DHEAD, NLM, DHEAD, 1.f, 0, 0, 0);
            // P1 = softmax(q @ k_l^T)  [16384, 256]
            gemm(qkvb + h * DHEAD, klb + (long)h * 16384, scor, nullptr, nullptr, nullptr,
                 NTOK, NLM, DHEAD, 1536, DHEAD, NLM, 0, 0, 0, 0, 1.f, 0, 1);
            k_softmax<<<NTOK, 256, 0, stream>>>(scor, scorb, nullptr, NLM);
            // out_h = P1 @ W -> attn[:, h*64 : h*64+64]
            gemm(scorb, WTb, attn + h * DHEAD, nullptr, nullptr, nullptr,
                 NTOK, DHEAD, NLM, NLM, NLM, CDIM, 0, 0, 0, 0, 1.f, 0, 1);
        }
        // residual depthwise conv on v (kernel 33) added into attn
        k_resconv<<<nb((long)NTOK * CDIM), 256, 0, stream>>>(qkvb, res_w[li], attn, NTOK);
        // h += attn @ out_w + out_b
        k_f32_to_bf16<<<nb((long)NTOK * CDIM), 256, 0, stream>>>(attn, xln, (long)NTOK * CDIM);
        k_transpose_f2b<<<nb((long)512 * 512), 256, 0, stream>>>(ao_w[li], aoT, 512, 512, 512, 1.f, 0, 0, 0);
        gemm(xln, aoT, hbuf, nullptr, ao_b[li], hbuf,
             NTOK, 512, 512, 512, 512, 512, 512, 0, 0, 0, 1.f, 0, 1);
    };

    layer(0);

    // ---- FPEG (between layers): h += x1 + x2 + x3 ----
    k_fpeg3<<<nb((long)NTOK * CDIM), 256, 0, stream>>>(hbuf, fpA,
        pj[0], pj[1], 7, pj[2], pj[3], 5, pj[4], pj[5], 7, 1);            // x1
    k_fpeg3<<<nb((long)NTOK * CDIM), 256, 0, stream>>>(fpA, fpB,
        pj[4], pj[5], 7, pj[6], pj[7], 5, pj[8], pj[9], 3, 1);            // x2
    k_add_f32<<<nb((long)NTOK * CDIM), 256, 0, stream>>>(hbuf, fpA, (long)NTOK * CDIM);
    k_fpeg3<<<nb((long)NTOK * CDIM), 256, 0, stream>>>(fpB, fpA,
        pj[10], pj[11], 7, pj[12], pj[13], 5, pj[14], pj[15], 3, 0);      // x3
    k_add_f32<<<nb((long)NTOK * CDIM), 256, 0, stream>>>(hbuf, fpB, (long)NTOK * CDIM);
    k_add_f32<<<nb((long)NTOK * CDIM), 256, 0, stream>>>(hbuf, fpA, (long)NTOK * CDIM);

    layer(1);

    // ---- head: LN -> fc2 -> argmax-select -> softmax ----
    k_layernorm<<<(NTOK + 3) / 4, 128, 0, stream>>>(hbuf, norm_w, norm_b, nullptr, hln, NTOK);
    k_fc2<<<(NTOK + 3) / 4, 128, 0, stream>>>(hln, fc2_w, fc2_b, scor, NTOK);
    k_final<<<1, 256, 0, stream>>>(scor, (float*)d_out, NTOK);
}
